// CausalSelfAttention_44384192037186
// MI455X (gfx1250) — matrix operand order repaired and checked by reading, not compiled, not run on hardware
//
#include <hip/hip_runtime.h>
#include <hip/hip_bf16.h>

// ---------------------------------------------------------------------------
// Causal self-attention w/ RoPE for MI455X (gfx1250, wave32, WMMA bf16 + TDM)
// B=4, T=2048, C=1024, H=16, D_HEAD=64
// ---------------------------------------------------------------------------

#define BB 4
#define TT 2048
#define CC 1024
#define HH 16
#define DH 64
#define MTOT (BB * TT)      // 8192
#define NQKV (3 * CC)       // 3072

typedef __bf16 bf16_t;
typedef __attribute__((ext_vector_type(16))) __bf16 v16bf;
typedef __attribute__((ext_vector_type(8)))  float  v8f;
typedef __attribute__((ext_vector_type(4)))  unsigned int v4u;
typedef __attribute__((ext_vector_type(8)))  int    v8i;
typedef __attribute__((ext_vector_type(4)))  int    v4i;

// Native f32 -> bf16 (RNE) via hardware convert; storage type is ushort.
__device__ __forceinline__ unsigned short f2bf(float f) {
    __bf16 h = (__bf16)f;
    return __builtin_bit_cast(unsigned short, h);
}
__device__ __forceinline__ unsigned int f2bf2(float lo, float hi) {
    return (unsigned int)f2bf(lo) | ((unsigned int)f2bf(hi) << 16);
}
__device__ __forceinline__ float bf2f(unsigned short h) {
    return (float)__builtin_bit_cast(__bf16, h);
}

union FragU { uint4 q[2]; v16bf v; };
union U4S  { uint4 u; unsigned short s[8]; };

// Build a 16-element bf16 fragment from two 16B-aligned runs of 8 bf16.
__device__ __forceinline__ v16bf make_frag(const unsigned short* p0,
                                           const unsigned short* p1) {
    FragU f;
    f.q[0] = *reinterpret_cast<const uint4*>(p0);
    f.q[1] = *reinterpret_cast<const uint4*>(p1);
    return f.v;
}

__device__ __forceinline__ v8f wmma_bf16(v16bf a, v16bf b, v8f c) {
    return __builtin_amdgcn_wmma_f32_16x16x32_bf16(false, a, false, b,
                                                   (short)0, c, false, false);
}

// ---------------------------------------------------------------------------
// Tensor Data Mover: 2D tile load (bf16) global -> LDS with row padding.
// Tile = tile_rows x 64 bf16, row stride in memory = 64 elems, LDS rows padded
// 128B data + 16B pad (pad_interval code 4 = 32 dwords, pad_amount code 3 = 4
// dwords) to match the [rows][72] ushort LDS layout.
// ---------------------------------------------------------------------------
#if __has_builtin(__builtin_amdgcn_tensor_load_to_lds) && \
    __has_builtin(__builtin_amdgcn_s_wait_tensorcnt)
#define USE_TDM 1
#else
#define USE_TDM 0
#endif

#if USE_TDM
__device__ __forceinline__ void tdm_load_tile_2d(const void* gptr,
                                                 unsigned lds_off,
                                                 int tile_rows) {
    unsigned long long ga = (unsigned long long)gptr;
    // Group0: count=1 | lds_addr | global_addr[56:0] | type=2 (bits 127:126)
    v4u g0 = { 1u,
               lds_off,
               (unsigned int)ga,
               ((unsigned int)(ga >> 32) & 0x01FFFFFFu) | 0x80000000u };
    // Group1: data_size=2B (17:16), pad_enable(20), pad_interval=4 (24:22),
    // pad_amount=3 (31:25); tensor_dim0=64, tensor_dim1=tile_rows,
    // tile_dim0=64, tile_dim1=tile_rows, tensor_dim0_stride=64.
    v8i g1 = { (int)0x07110000,
               64 << 16,                    // tensor_dim0[15:0] -> [31:16]
               (tile_rows & 0xFFFF) << 16,  // tensor_dim1[15:0] -> [31:16]
               64 << 16,                    // tile_dim0 -> [31:16]
               tile_rows,                   // tile_dim1
               64,                          // tensor_dim0_stride[31:0]
               0, 0 };
    v4i gz4 = { 0, 0, 0, 0 };              // groups 2/3: dims unused (2D)
    v8i gz8 = { 0, 0, 0, 0, 0, 0, 0, 0 };  // extension group: unused
    __builtin_amdgcn_tensor_load_to_lds(g0, g1, gz4, gz4, gz8, 0);
}
#endif

__device__ __forceinline__ unsigned lds_off_of(const void* p) {
    return (unsigned)(unsigned long long)p;   // low 32 bits = LDS byte offset
}

// ---------------------------------------------------------------------------
// Kernel 1: qkv = x @ W_qkv ; scatter into Q/K/V bf16 buffers [B,H,T,D]
// ---------------------------------------------------------------------------
__global__ __launch_bounds__(256) void qkv_gemm_kernel(
    const float* __restrict__ x, const float* __restrict__ Wqkv,
    unsigned short* __restrict__ Qb, unsigned short* __restrict__ Kb,
    unsigned short* __restrict__ Vb) {

    __shared__ __align__(16) unsigned short As[64][40];  // [m][k], 80B rows
    __shared__ __align__(16) unsigned short Bs[64][40];  // [n][k] (transposed)

    const int tid  = threadIdx.x;
    const int lane = tid & 31;
    const int wave = tid >> 5;
    const int half = lane >> 4;
    const int ln   = lane & 15;
    const int wm   = (wave >> 1) * 16;
    const int wn   = (wave & 1) * 32;

    const int n0 = blockIdx.x * 64;
    const int m0 = blockIdx.y * 64;

    v8f acc0 = {0.f,0.f,0.f,0.f,0.f,0.f,0.f,0.f};
    v8f acc1 = {0.f,0.f,0.f,0.f,0.f,0.f,0.f,0.f};

    const int ra = tid >> 3, ca = (tid & 7) * 4;     // A-fill coords
    const int rb = tid >> 4, cb = (tid & 15) * 4;    // B-fill coords

    for (int k0 = 0; k0 < CC; k0 += 32) {
        // A: 64 rows x 32 k, float4 loads, packed bf16x2 stores (2 rows/thread)
        #pragma unroll
        for (int it = 0; it < 2; ++it) {
            int r = ra + it * 32;
            float4 f = *reinterpret_cast<const float4*>(x + (m0 + r) * CC + k0 + ca);
            uint2 p; p.x = f2bf2(f.x, f.y); p.y = f2bf2(f.z, f.w);
            *reinterpret_cast<uint2*>(&As[r][ca]) = p;
        }
        // B: 32 k x 64 n -> [n][k] transposed scatter (2 k-rows/thread)
        #pragma unroll
        for (int it = 0; it < 2; ++it) {
            int r = rb + it * 16;
            float4 f = *reinterpret_cast<const float4*>(Wqkv + (k0 + r) * NQKV + n0 + cb);
            Bs[cb + 0][r] = f2bf(f.x);
            Bs[cb + 1][r] = f2bf(f.y);
            Bs[cb + 2][r] = f2bf(f.z);
            Bs[cb + 3][r] = f2bf(f.w);
        }
        __syncthreads();

        v16bf a  = make_frag(&As[wm + ln][16 * half],
                             &As[wm + ln][16 * half + 8]);
        v16bf b0 = make_frag(&Bs[wn + ln][16 * half],
                             &Bs[wn + ln][16 * half + 8]);
        v16bf b1 = make_frag(&Bs[wn + 16 + ln][16 * half],
                             &Bs[wn + 16 + ln][16 * half + 8]);
        acc0 = wmma_bf16(a, b0, acc0);
        acc1 = wmma_bf16(a, b1, acc1);
        __syncthreads();
    }

    #pragma unroll
    for (int tile = 0; tile < 2; ++tile) {
        const int   gn     = n0 + wn + tile * 16 + ln;
        const int   which  = gn >> 10;          // 0=Q 1=K 2=V
        const int   within = gn & 1023;
        const int   h      = within >> 6;
        const int   d      = within & 63;
        unsigned short* dst = (which == 0) ? Qb : (which == 1) ? Kb : Vb;
        #pragma unroll
        for (int r = 0; r < 8; ++r) {
            const int gm = m0 + wm + r + 8 * half;
            const int b  = gm >> 11;
            const int t  = gm & 2047;
            const float v = (tile == 0) ? acc0[r] : acc1[r];
            dst[(((b * HH + h) * TT + t) << 6) + d] = f2bf(v);
        }
    }
}

// ---------------------------------------------------------------------------
// Kernel 2: RoPE in place on Q,K (bf16). Folds softmax scale (0.125) into Q.
// ---------------------------------------------------------------------------
__global__ __launch_bounds__(256) void rope_kernel(
    unsigned short* __restrict__ Qb, unsigned short* __restrict__ Kb) {
    const int idx = blockIdx.x * blockDim.x + threadIdx.x;
    if (idx >= BB * HH * TT * 32) return;
    const int d  = idx & 31;
    const int t  = (idx >> 5) & 2047;
    const int bh = idx >> 16;
    const int base = (bh * TT + t) * DH;

    const float inv = __expf(-(float)d * 0.28782313662425572f); // ln(1e4)/32
    float s, c;
    __sincosf((float)t * inv, &s, &c);

    {
        float x0 = bf2f(Qb[base + d]);
        float x1 = bf2f(Qb[base + d + 32]);
        Qb[base + d]      = f2bf((x0 * c - x1 * s) * 0.125f);
        Qb[base + d + 32] = f2bf((x1 * c + x0 * s) * 0.125f);
    }
    {
        float x0 = bf2f(Kb[base + d]);
        float x1 = bf2f(Kb[base + d + 32]);
        Kb[base + d]      = f2bf(x0 * c - x1 * s);
        Kb[base + d + 32] = f2bf(x1 * c + x0 * s);
    }
}

// ---------------------------------------------------------------------------
// Kernel 3: flash attention. 256 threads / 8 waves handle 128 queries of one
// (b,h). Q and K tiles staged by the Tensor Data Mover; V staged manually
// (needs [d][key] transpose). Online softmax is wave-local (16 rows/wave).
// ---------------------------------------------------------------------------
__global__ __launch_bounds__(256) void flash_kernel(
    const unsigned short* __restrict__ Qb, const unsigned short* __restrict__ Kb,
    const unsigned short* __restrict__ Vb, unsigned short* __restrict__ att) {

    __shared__ __align__(16) unsigned short Qs[128][72]; // [q][d]
    __shared__ __align__(16) unsigned short Ks[64][72];  // [key][d]
    __shared__ __align__(16) unsigned short Vt[64][72];  // [d][key]
    __shared__ __align__(16) unsigned short Ps[8][16][72]; // per-wave P 16x64

    const int tid  = threadIdx.x;
    const int lane = tid & 31;
    const int wave = tid >> 5;
    const int half = lane >> 4;
    const int ln   = lane & 15;

    const int qblk = blockIdx.x;
    const int bh   = blockIdx.y;
    const int qb0  = qblk * 128;
    const int base_bh = bh * (TT * DH);

#if USE_TDM
    if (wave == 0)
        tdm_load_tile_2d(Qb + base_bh + qb0 * DH, lds_off_of(&Qs[0][0]), 128);
#else
    #pragma unroll
    for (int it = 0; it < 4; ++it) {
        int r = (tid >> 3) + it * 32, c8 = (tid & 7) * 8;
        *reinterpret_cast<uint4*>(&Qs[r][c8]) =
            *reinterpret_cast<const uint4*>(Qb + base_bh + (qb0 + r) * DH + c8);
    }
#endif

    float mrow[8], lrow[8];
    v8f o[4];
    #pragma unroll
    for (int r = 0; r < 8; ++r) { mrow[r] = -1e30f; lrow[r] = 0.f; }
    #pragma unroll
    for (int dt = 0; dt < 4; ++dt) o[dt] = (v8f){0.f,0.f,0.f,0.f,0.f,0.f,0.f,0.f};

    const int rv = tid >> 3, cv = (tid & 7) * 8;     // V-fill coords

    const int nblk = 2 * qblk + 2;
    for (int jb = 0; jb < nblk; ++jb) {
        __syncthreads();                // previous iteration's LDS reads done
#if USE_TDM
        if (wave == 0)
            tdm_load_tile_2d(Kb + base_bh + jb * 64 * DH,
                             lds_off_of(&Ks[0][0]), 64);
#else
        #pragma unroll
        for (int it = 0; it < 2; ++it) {
            int r = rv + it * 32;
            *reinterpret_cast<uint4*>(&Ks[r][cv]) =
                *reinterpret_cast<const uint4*>(Kb + base_bh + (jb * 64 + r) * DH + cv);
        }
#endif
        // V transposed fill (uint4 global loads, ushort LDS scatter)
        #pragma unroll
        for (int it = 0; it < 2; ++it) {
            int key = rv + it * 32;
            U4S t;
            t.u = *reinterpret_cast<const uint4*>(Vb + base_bh + (jb * 64 + key) * DH + cv);
            #pragma unroll
            for (int j = 0; j < 8; ++j) Vt[cv + j][key] = t.s[j];
        }
#if USE_TDM
        if (wave == 0) __builtin_amdgcn_s_wait_tensorcnt(0);
#endif
        __syncthreads();

        // ---- S = Q Kt : 16 queries x 64 keys per wave --------------------
        v8f s[4];
        #pragma unroll
        for (int nt = 0; nt < 4; ++nt) s[nt] = (v8f){0.f,0.f,0.f,0.f,0.f,0.f,0.f,0.f};
        #pragma unroll
        for (int kk = 0; kk < 64; kk += 32) {
            v16bf a = make_frag(&Qs[wave * 16 + ln][kk + 16 * half],
                                &Qs[wave * 16 + ln][kk + 16 * half + 8]);
            #pragma unroll
            for (int nt = 0; nt < 4; ++nt) {
                v16bf b = make_frag(&Ks[nt * 16 + ln][kk + 16 * half],
                                    &Ks[nt * 16 + ln][kk + 16 * half + 8]);
                s[nt] = wmma_bf16(a, b, s[nt]);
            }
        }

        // ---- causal mask + online softmax --------------------------------
        #pragma unroll
        for (int r = 0; r < 8; ++r) {
            const int q = qb0 + wave * 16 + r + 8 * half;
            float pv[4];
            float rowmax = -1e30f;
            #pragma unroll
            for (int nt = 0; nt < 4; ++nt) {
                const int key = jb * 64 + nt * 16 + ln;
                float v = (key <= q) ? s[nt][r] : -1e30f;
                pv[nt] = v;
                rowmax = fmaxf(rowmax, v);
            }
            #pragma unroll
            for (int off = 1; off < 16; off <<= 1)
                rowmax = fmaxf(rowmax, __shfl_xor(rowmax, off, 32));
            const float mnew = fmaxf(mrow[r], rowmax);
            const float corr = __expf(mrow[r] - mnew);
            mrow[r] = mnew;
            float rsum = 0.f;
            #pragma unroll
            for (int nt = 0; nt < 4; ++nt) {
                float p = __expf(pv[nt] - mnew);
                rsum += p;
                Ps[wave][r + 8 * half][nt * 16 + ln] = f2bf(p);
            }
            #pragma unroll
            for (int off = 1; off < 16; off <<= 1)
                rsum += __shfl_xor(rsum, off, 32);
            lrow[r] = lrow[r] * corr + rsum;
            #pragma unroll
            for (int dt = 0; dt < 4; ++dt) o[dt][r] *= corr;
        }
        __syncthreads();

        // ---- O += P V -----------------------------------------------------
        #pragma unroll
        for (int kk = 0; kk < 64; kk += 32) {
            v16bf a = make_frag(&Ps[wave][ln][kk + 16 * half],
                                &Ps[wave][ln][kk + 16 * half + 8]);
            #pragma unroll
            for (int dt = 0; dt < 4; ++dt) {
                v16bf b = make_frag(&Vt[dt * 16 + ln][kk + 16 * half],
                                    &Vt[dt * 16 + ln][kk + 16 * half + 8]);
                o[dt] = wmma_bf16(a, b, o[dt]);
            }
        }
    }

    // ---- normalize + store att[b, t, h*64+d] (bf16) ----------------------
    const int bidx = bh >> 4;
    const int h    = bh & 15;
    #pragma unroll
    for (int dt = 0; dt < 4; ++dt) {
        #pragma unroll
        for (int r = 0; r < 8; ++r) {
            const int q = qb0 + wave * 16 + r + 8 * half;
            const int d = dt * 16 + ln;
            att[(bidx * TT + q) * CC + h * 64 + d] = f2bf(o[dt][r] / lrow[r]);
        }
    }
}

// ---------------------------------------------------------------------------
// Kernel 4: out = att @ W_proj  (bf16 A, bf16-converted B, f32 out)
// ---------------------------------------------------------------------------
__global__ __launch_bounds__(256) void proj_gemm_kernel(
    const unsigned short* __restrict__ attA, const float* __restrict__ Wp,
    float* __restrict__ out) {

    __shared__ __align__(16) unsigned short As[64][40];
    __shared__ __align__(16) unsigned short Bs[64][40];  // [n][k]

    const int tid  = threadIdx.x;
    const int lane = tid & 31;
    const int wave = tid >> 5;
    const int half = lane >> 4;
    const int ln   = lane & 15;
    const int wm   = (wave >> 1) * 16;
    const int wn   = (wave & 1) * 32;

    const int n0 = blockIdx.x * 64;
    const int m0 = blockIdx.y * 64;

    v8f acc0 = {0.f,0.f,0.f,0.f,0.f,0.f,0.f,0.f};
    v8f acc1 = {0.f,0.f,0.f,0.f,0.f,0.f,0.f,0.f};

    const int ra = tid >> 2, ca = (tid & 3) * 8;     // A-fill coords
    const int rb = tid >> 4, cb = (tid & 15) * 4;    // B-fill coords

    for (int k0 = 0; k0 < CC; k0 += 32) {
        {   // A: bf16 pass-through, one uint4 per thread
            *reinterpret_cast<uint4*>(&As[ra][ca]) =
                *reinterpret_cast<const uint4*>(attA + (m0 + ra) * CC + k0 + ca);
        }
        #pragma unroll
        for (int it = 0; it < 2; ++it) {
            int r = rb + it * 16;
            float4 f = *reinterpret_cast<const float4*>(Wp + (k0 + r) * CC + n0 + cb);
            Bs[cb + 0][r] = f2bf(f.x);
            Bs[cb + 1][r] = f2bf(f.y);
            Bs[cb + 2][r] = f2bf(f.z);
            Bs[cb + 3][r] = f2bf(f.w);
        }
        __syncthreads();

        v16bf a  = make_frag(&As[wm + ln][16 * half],
                             &As[wm + ln][16 * half + 8]);
        v16bf b0 = make_frag(&Bs[wn + ln][16 * half],
                             &Bs[wn + ln][16 * half + 8]);
        v16bf b1 = make_frag(&Bs[wn + 16 + ln][16 * half],
                             &Bs[wn + 16 + ln][16 * half + 8]);
        acc0 = wmma_bf16(a, b0, acc0);
        acc1 = wmma_bf16(a, b1, acc1);
        __syncthreads();
    }

    #pragma unroll
    for (int tile = 0; tile < 2; ++tile) {
        const int gn = n0 + wn + tile * 16 + ln;
        #pragma unroll
        for (int r = 0; r < 8; ++r) {
            const int gm = m0 + wm + r + 8 * half;
            out[gm * CC + gn] = (tile == 0) ? acc0[r] : acc1[r];
        }
    }
}

// ---------------------------------------------------------------------------
extern "C" void kernel_launch(void* const* d_in, const int* in_sizes, int n_in,
                              void* d_out, int out_size, void* d_ws, size_t ws_size,
                              hipStream_t stream) {
    const float* x     = (const float*)d_in[0];   // [4,2048,1024]
    const float* Wqkv  = (const float*)d_in[1];   // [1024,3072]
    const float* Wproj = (const float*)d_in[2];   // [1024,1024]
    float*       out   = (float*)d_out;           // [4,2048,1024]

    const size_t elems = (size_t)BB * HH * TT * DH;   // 8M
    unsigned short* Qb  = (unsigned short*)d_ws;
    unsigned short* Kb  = Qb + elems;
    unsigned short* Vb  = Kb + elems;
    unsigned short* att = Vb + elems;                 // 8192*1024

    qkv_gemm_kernel<<<dim3(NQKV / 64, MTOT / 64), 256, 0, stream>>>(
        x, Wqkv, Qb, Kb, Vb);

    rope_kernel<<<(BB * HH * TT * 32) / 256, 256, 0, stream>>>(Qb, Kb);

    flash_kernel<<<dim3(TT / 128, BB * HH), 256, 0, stream>>>(Qb, Kb, Vb, att);

    proj_gemm_kernel<<<dim3(CC / 64, MTOT / 64), 256, 0, stream>>>(
        att, Wproj, out);
}
